// RXFOOD_partial_2843268350074
// MI455X (gfx1250) — matrix operand order repaired and checked by reading, not compiled, not run on hardware
//
#include <hip/hip_runtime.h>
#include <hip/hip_bf16.h>

// ---------------------------------------------------------------------------
// RXFOOD cross-domain channel attention, MI455X (gfx1250) wave32 + WMMA bf16.
// All GEMMs run on v_wmma_f32_16x16x32_bf16 (f32 accumulate), with
// register-pipelined global->LDS staging (b128/b64 loads, b64 DS stores).
// ---------------------------------------------------------------------------

typedef __attribute__((ext_vector_type(16))) __bf16 v16bf;
typedef __attribute__((ext_vector_type(8)))  float  v8f;

#define NBATCH 8

// Epilogue modes
#define EPI_PROJ 0   // bf16 out = acc + bias[m]
#define EPI_ENER 1   // f32  out = acc * alpha
#define EPI_AV   2   // bf16 out = acc
#define EPI_OUT  3   // f32  out = g[d]*(acc + rb[m]) + 2*x[m][n]

union BF4 {
  __bf16 h[4];
  unsigned long long u;
};

// Batched GEMM: D[z] = A[z] * B[z] (+epilogue), z = d*8 + b over 16 batches.
// Block tile 128(M) x 64(N), K-step 32. 256 threads = 8 wave32:
//   wave w: M-strip = w*16, full 64 N -> four 16x16 accumulators (one a-frag
//   feeds 4 WMMAs). Tile t+1 is prefetched to registers during compute of t.
template <int EPI, bool ABF, bool BBF, bool BT>
__global__ __launch_bounds__(256) void gemm_wmma(
    const void* __restrict__ Aall,
    const void* __restrict__ Bd0, const void* __restrict__ Bd1,
    void* __restrict__ Dst,
    const float* __restrict__ bias,   // per-d: bias + d*M (EPI_PROJ / EPI_OUT)
    const float* __restrict__ x0, const float* __restrict__ x1,  // EPI_OUT
    const float* __restrict__ gvec,   // EPI_OUT gamma[2]
    float alpha,
    int M, int N, int K, int lda, int ldb,
    long long aStrideD, long long aStrideZ,
    long long bStrideB, long long bStrideZ,
    long long dStrideZ, long long dStrideD, long long dStrideB,
    long long xStrideB)
{
  const int z = blockIdx.z;
  const int d = z >> 3;
  const int b = z & 7;

  const int m0 = blockIdx.x * 128;
  const int n0 = blockIdx.y * 64;

  const long long aoff = (long long)d * aStrideD + (long long)z * aStrideZ;
  const float*  Af = (const float*)Aall + aoff;
  const __bf16* Ab = (const __bf16*)Aall + aoff;

  const void* Bbase = d ? Bd1 : Bd0;
  const long long boff = (long long)z * bStrideZ + (long long)b * bStrideB;
  const float*  Bf = (const float*)Bbase + boff;
  const __bf16* Bb = (const __bf16*)Bbase + boff;

  const long long doff = (long long)z * dStrideZ + (long long)d * dStrideD +
                         (long long)b * dStrideB;

  // Row strides are 16-byte multiples so fragment reads stay ds_load_b128.
  __shared__ __bf16 As[128][48];   // 128 x 32 (+pad), 96 B rows
  __shared__ __bf16 Bs[32][72];    // 32 x 64 (+pad), 144 B rows

  const int tid  = threadIdx.x;
  const int wave = tid >> 5;       // 0..7 -> M strip
  const int lane = tid & 31;
  const int half = lane >> 4;
  const int mr   = lane & 15;

  // Staging registers (tile double-buffer lives in registers, single LDS buf).
  float4 ra_f[4]; unsigned long long ra_b[4];
  float4 rb_f[2]; unsigned long long rb_b[2];

  auto loadT = [&](int k0) {
    // A tile: 128x32 = 4096 elems, 4-elem chunks, 4 per thread.
#pragma unroll
    for (int j = 0; j < 4; ++j) {
      int cidx = tid + j * 256;
      int r = cidx >> 3, c = (cidx & 7) * 4;
      long long src = (long long)(m0 + r) * lda + (k0 + c);
      if constexpr (ABF) ra_b[j] = *(const unsigned long long*)(Ab + src);
      else               ra_f[j] = *(const float4*)(Af + src);
    }
    // B tile: 32x64 = 2048 elems, 4-elem chunks, 2 per thread.
#pragma unroll
    for (int j = 0; j < 2; ++j) {
      int cidx = tid + j * 256;
      if constexpr (BT) {
        int kk = (cidx & 7) * 4, n = cidx >> 3;
        long long src = (long long)(n0 + n) * ldb + (k0 + kk);
        rb_b[j] = *(const unsigned long long*)(Bb + src);
      } else {
        int r = cidx >> 4, c = (cidx & 15) * 4;
        long long src = (long long)(k0 + r) * ldb + (n0 + c);
        if constexpr (BBF) rb_b[j] = *(const unsigned long long*)(Bb + src);
        else               rb_f[j] = *(const float4*)(Bf + src);
      }
    }
  };

  auto storeT = [&]() {
#pragma unroll
    for (int j = 0; j < 4; ++j) {
      int cidx = tid + j * 256;
      int r = cidx >> 3, c = (cidx & 7) * 4;
      if constexpr (ABF) {
        *(unsigned long long*)&As[r][c] = ra_b[j];
      } else {
        BF4 t;
        t.h[0] = (__bf16)ra_f[j].x; t.h[1] = (__bf16)ra_f[j].y;
        t.h[2] = (__bf16)ra_f[j].z; t.h[3] = (__bf16)ra_f[j].w;
        *(unsigned long long*)&As[r][c] = t.u;
      }
    }
#pragma unroll
    for (int j = 0; j < 2; ++j) {
      int cidx = tid + j * 256;
      if constexpr (BT) {
        int kk = (cidx & 7) * 4, n = cidx >> 3;
        BF4 t; t.u = rb_b[j];
        Bs[kk + 0][n] = t.h[0]; Bs[kk + 1][n] = t.h[1];
        Bs[kk + 2][n] = t.h[2]; Bs[kk + 3][n] = t.h[3];
      } else {
        int r = cidx >> 4, c = (cidx & 15) * 4;
        if constexpr (BBF) {
          *(unsigned long long*)&Bs[r][c] = rb_b[j];
        } else {
          BF4 t;
          t.h[0] = (__bf16)rb_f[j].x; t.h[1] = (__bf16)rb_f[j].y;
          t.h[2] = (__bf16)rb_f[j].z; t.h[3] = (__bf16)rb_f[j].w;
          *(unsigned long long*)&Bs[r][c] = t.u;
        }
      }
    }
  };

  v8f acc[4];
#pragma unroll
  for (int s = 0; s < 4; ++s) acc[s] = (v8f){};

  const int T = K >> 5;
  loadT(0);
  for (int t = 0; t < T; ++t) {
    storeT();                       // waits on tile-t global loads here
    __syncthreads();
    if (t + 1 < T) loadT((t + 1) << 5);  // in flight during WMMAs below

    // A fragment (ISA 7.12.2): lane m=mr, halves split K
    v16bf a;
#pragma unroll
    for (int e = 0; e < 8; ++e) {
      a[e]     = As[wave * 16 + mr][half * 8 + e];
      a[e + 8] = As[wave * 16 + mr][16 + half * 8 + e];
    }
#pragma unroll
    for (int s = 0; s < 4; ++s) {
      v16bf bb;
#pragma unroll
      for (int e = 0; e < 8; ++e) { bb[e] = Bs[half * 8 + e][s * 16 + mr]; bb[e + 8] = Bs[16 + half * 8 + e][s * 16 + mr]; }
      acc[s] = __builtin_amdgcn_wmma_f32_16x16x32_bf16(
          false, a, false, bb, (short)0, acc[s], false, false);
    }
    __syncthreads();
  }

  // --- epilogue / store.  D layout: VGPR r -> M = 8*half + r, lane -> N ---
  const int nn = lane & 15;
  float g = 0.0f;
  const float* xr = nullptr;
  if constexpr (EPI == EPI_OUT) {
    g  = gvec[d];
    xr = (d ? x1 : x0) + (long long)b * xStrideB;
  }
  const float* biasp = bias ? (bias + (long long)d * M) : nullptr;

#pragma unroll
  for (int s = 0; s < 4; ++s) {
    int gn = n0 + s * 16 + nn;
#pragma unroll
    for (int r = 0; r < 8; ++r) {
      int gm = m0 + wave * 16 + 8 * half + r;
      float v = acc[s][r];
      long long di = doff + (long long)gm * N + gn;
      if constexpr (EPI == EPI_PROJ) {
        ((__bf16*)Dst)[di] = (__bf16)(v + biasp[gm]);
      } else if constexpr (EPI == EPI_ENER) {
        ((float*)Dst)[di] = v * alpha;
      } else if constexpr (EPI == EPI_AV) {
        ((__bf16*)Dst)[di] = (__bf16)v;
      } else {  // EPI_OUT
        float xv = xr[(long long)gm * N + gn];
        ((float*)Dst)[di] = g * (v + biasp[gm]) + 2.0f * xv;
      }
    }
  }
}

// ---------------------------------------------------------------------------
// Bilinear align-corners helper (square n x n source, non-negative coords)
// ---------------------------------------------------------------------------
__device__ __forceinline__ float bilin_ac(const float* __restrict__ p, int n,
                                          float cy, float cx) {
  int y0 = (int)cy; if (y0 > n - 1) y0 = n - 1;
  int y1 = y0 + 1;  if (y1 > n - 1) y1 = n - 1;
  float wy = cy - (float)y0;
  int x0 = (int)cx; if (x0 > n - 1) x0 = n - 1;
  int x1 = x0 + 1;  if (x1 > n - 1) x1 = n - 1;
  float wx = cx - (float)x0;
  float top = p[y0 * n + x0] * (1.0f - wx) + p[y0 * n + x1] * wx;
  float bot = p[y1 * n + x0] * (1.0f - wx) + p[y1 * n + x1] * wx;
  return top * (1.0f - wy) + bot * wy;
}

// energies[2,B,ci,ci] -> enermap[B,4,256,256] channel (2*d + j), resized
__global__ void resize_to_enermap(const float* __restrict__ E,
                                  float* __restrict__ EN, int ci, int j) {
  int idx = blockIdx.x * blockDim.x + threadIdx.x;
  if (idx >= 2 * NBATCH * 256 * 256) return;
  int ox = idx & 255;
  int oy = (idx >> 8) & 255;
  int b  = (idx >> 16) & 7;
  int d  = idx >> 19;
  float s  = (float)(ci - 1) / 255.0f;
  const float* e = E + ((long long)(d * NBATCH + b)) * ci * ci;
  float val = bilin_ac(e, ci, oy * s, ox * s);
  int ch = 2 * d + j;
  EN[(((long long)b * 4 + ch) * 256 + oy) * 256 + ox] = val;
}

// 4x4 1x1 conv over enermap channels
__global__ void energy_conv(const float* __restrict__ EN, float* __restrict__ EN2,
                            const float* __restrict__ ew,
                            const float* __restrict__ eb) {
  int idx = blockIdx.x * blockDim.x + threadIdx.x;
  if (idx >= NBATCH * 256 * 256) return;
  int p = idx & 65535;
  int b = idx >> 16;
  long long base = (long long)b * 4 * 65536 + p;
  float c0 = EN[base], c1 = EN[base + 65536];
  float c2 = EN[base + 2 * 65536], c3 = EN[base + 3 * 65536];
#pragma unroll
  for (int o = 0; o < 4; ++o) {
    EN2[base + (long long)o * 65536] =
        ew[o * 4 + 0] * c0 + ew[o * 4 + 1] * c1 +
        ew[o * 4 + 2] * c2 + ew[o * 4 + 3] * c3 + eb[o];
  }
}

// Resize enermap channel back to ci x ci and row-softmax -> bf16 attention.
// One block per output row; blockDim.x == ci (128 or 256, power of two).
__global__ void resize_softmax(const float* __restrict__ EN2,
                               __bf16* __restrict__ att, int ci, int j) {
  int row = blockIdx.x;            // over 2*B*ci
  int i = row % ci;
  int b = (row / ci) & 7;
  int d = row / (ci * NBATCH);
  int c = threadIdx.x;

  float s = 255.0f / (float)(ci - 1);
  int ch = 2 * d + j;
  const float* src = EN2 + ((long long)b * 4 + ch) * 65536;
  float v = bilin_ac(src, 256, i * s, c * s);

  __shared__ float red[256];
  red[c] = v;
  __syncthreads();
  for (int off = ci >> 1; off > 0; off >>= 1) {
    if (c < off) red[c] = fmaxf(red[c], red[c + off]);
    __syncthreads();
  }
  float mx = red[0];
  __syncthreads();
  float ex = __expf(v - mx);
  red[c] = ex;
  __syncthreads();
  for (int off = ci >> 1; off > 0; off >>= 1) {
    if (c < off) red[c] += red[c + off];
    __syncthreads();
  }
  float inv = 1.0f / red[0];
  att[((long long)(d * NBATCH + b) * ci + i) * ci + c] = (__bf16)(ex * inv);
}

// ---------------------------------------------------------------------------
extern "C" void kernel_launch(void* const* d_in, const int* in_sizes, int n_in,
                              void* d_out, int out_size, void* d_ws, size_t ws_size,
                              hipStream_t stream) {
  (void)in_sizes; (void)n_in; (void)out_size; (void)ws_size;
  const float* rgb0  = (const float*)d_in[0];
  const float* rgb1  = (const float*)d_in[1];
  const float* freq0 = (const float*)d_in[2];
  const float* freq1 = (const float*)d_in[3];
  const float* qw0 = (const float*)d_in[4],  *qb0 = (const float*)d_in[5];
  const float* kw0 = (const float*)d_in[6],  *kb0 = (const float*)d_in[7];
  const float* vw0 = (const float*)d_in[8],  *vb0 = (const float*)d_in[9];
  const float* rw0 = (const float*)d_in[10], *rb0 = (const float*)d_in[11];
  const float* g0  = (const float*)d_in[12];
  const float* qw1 = (const float*)d_in[13], *qb1 = (const float*)d_in[14];
  const float* kw1 = (const float*)d_in[15], *kb1 = (const float*)d_in[16];
  const float* vw1 = (const float*)d_in[17], *vb1 = (const float*)d_in[18];
  const float* rw1 = (const float*)d_in[19], *rb1 = (const float*)d_in[20];
  const float* g1  = (const float*)d_in[21];
  const float* ew  = (const float*)d_in[22], *eb = (const float*)d_in[23];

  // Geometry
  const int C0 = 256, CI0 = 128, HW0 = 4096;
  const int C1 = 512, CI1 = 256, HW1 = 1024;

  // Workspace carve (256B aligned). Total ~100 MB.
  char* ws = (char*)d_ws;
  size_t off = 0;
  auto carve = [&](size_t bytes) -> char* {
    char* p = ws + off;
    off += (bytes + 255) & ~(size_t)255;
    return p;
  };
  const size_t qkv0 = (size_t)2 * NBATCH * CI0 * HW0;  // elems
  const size_t qkv1 = (size_t)2 * NBATCH * CI1 * HW1;
  __bf16* Q0 = (__bf16*)carve(qkv0 * 2);
  __bf16* K0 = (__bf16*)carve(qkv0 * 2);
  __bf16* V0 = (__bf16*)carve(qkv0 * 2);
  __bf16* Q1 = (__bf16*)carve(qkv1 * 2);
  __bf16* K1 = (__bf16*)carve(qkv1 * 2);
  __bf16* V1 = (__bf16*)carve(qkv1 * 2);
  float*  E0 = (float*)carve((size_t)2 * NBATCH * CI0 * CI0 * 4);
  float*  E1 = (float*)carve((size_t)2 * NBATCH * CI1 * CI1 * 4);
  float*  EN1 = (float*)carve((size_t)NBATCH * 4 * 256 * 256 * 4);
  float*  EN2 = (float*)carve((size_t)NBATCH * 4 * 256 * 256 * 4);
  __bf16* ATT0 = (__bf16*)carve((size_t)2 * NBATCH * CI0 * CI0 * 2);
  __bf16* ATT1 = (__bf16*)carve((size_t)2 * NBATCH * CI1 * CI1 * 2);
  __bf16* O0 = Q0;  // stream-ordered alias: Q0 dead after energy0
  __bf16* O1 = Q1;  // stream-ordered alias: Q1 dead after energy1

  const dim3 blk(256);

  // ---- 1) Q/K/V projections (A=f32 weights, B=f32 inputs -> bf16) ----
  {
    dim3 g0d(CI0 / 128, HW0 / 64, 16);
    gemm_wmma<EPI_PROJ, false, false, false><<<g0d, blk, 0, stream>>>(
        qw0, rgb0, freq0, Q0, qb0, nullptr, nullptr, nullptr, 1.0f,
        CI0, HW0, C0, C0, HW0,
        (long long)CI0 * C0, 0, (long long)C0 * HW0, 0,
        (long long)CI0 * HW0, 0, 0, 0);
    gemm_wmma<EPI_PROJ, false, false, false><<<g0d, blk, 0, stream>>>(
        kw0, rgb0, freq0, K0, kb0, nullptr, nullptr, nullptr, 1.0f,
        CI0, HW0, C0, C0, HW0,
        (long long)CI0 * C0, 0, (long long)C0 * HW0, 0,
        (long long)CI0 * HW0, 0, 0, 0);
    gemm_wmma<EPI_PROJ, false, false, false><<<g0d, blk, 0, stream>>>(
        vw0, rgb0, freq0, V0, vb0, nullptr, nullptr, nullptr, 1.0f,
        CI0, HW0, C0, C0, HW0,
        (long long)CI0 * C0, 0, (long long)C0 * HW0, 0,
        (long long)CI0 * HW0, 0, 0, 0);

    dim3 g1d(CI1 / 128, HW1 / 64, 16);
    gemm_wmma<EPI_PROJ, false, false, false><<<g1d, blk, 0, stream>>>(
        qw1, rgb1, freq1, Q1, qb1, nullptr, nullptr, nullptr, 1.0f,
        CI1, HW1, C1, C1, HW1,
        (long long)CI1 * C1, 0, (long long)C1 * HW1, 0,
        (long long)CI1 * HW1, 0, 0, 0);
    gemm_wmma<EPI_PROJ, false, false, false><<<g1d, blk, 0, stream>>>(
        kw1, rgb1, freq1, K1, kb1, nullptr, nullptr, nullptr, 1.0f,
        CI1, HW1, C1, C1, HW1,
        (long long)CI1 * C1, 0, (long long)C1 * HW1, 0,
        (long long)CI1 * HW1, 0, 0, 0);
    gemm_wmma<EPI_PROJ, false, false, false><<<g1d, blk, 0, stream>>>(
        vw1, rgb1, freq1, V1, vb1, nullptr, nullptr, nullptr, 1.0f,
        CI1, HW1, C1, C1, HW1,
        (long long)CI1 * C1, 0, (long long)C1 * HW1, 0,
        (long long)CI1 * HW1, 0, 0, 0);
  }

  // ---- 2) Energies: E = alpha * Q * K^T  (bf16 x bf16^T -> f32) ----
  gemm_wmma<EPI_ENER, true, true, true>
      <<<dim3(CI0 / 128, CI0 / 64, 16), blk, 0, stream>>>(
      Q0, K0, K0, E0, nullptr, nullptr, nullptr, nullptr, 0.08838834764831845f,
      CI0, CI0, HW0, HW0, HW0,
      0, (long long)CI0 * HW0, 0, (long long)CI0 * HW0,
      (long long)CI0 * CI0, 0, 0, 0);
  gemm_wmma<EPI_ENER, true, true, true>
      <<<dim3(CI1 / 128, CI1 / 64, 16), blk, 0, stream>>>(
      Q1, K1, K1, E1, nullptr, nullptr, nullptr, nullptr, 0.0625f,
      CI1, CI1, HW1, HW1, HW1,
      0, (long long)CI1 * HW1, 0, (long long)CI1 * HW1,
      (long long)CI1 * CI1, 0, 0, 0);

  // ---- 3) Resize energies to 256x256, fuse with 4x4 conv ----
  {
    int tot = 2 * NBATCH * 256 * 256;
    resize_to_enermap<<<(tot + 255) / 256, 256, 0, stream>>>(E0, EN1, CI0, 0);
    resize_to_enermap<<<(tot + 255) / 256, 256, 0, stream>>>(E1, EN1, CI1, 1);
    int tot2 = NBATCH * 256 * 256;
    energy_conv<<<(tot2 + 255) / 256, 256, 0, stream>>>(EN1, EN2, ew, eb);
  }

  // ---- 4) Resize back + row softmax -> bf16 attention ----
  resize_softmax<<<2 * NBATCH * CI0, CI0, 0, stream>>>(EN2, ATT0, CI0, 0);
  resize_softmax<<<2 * NBATCH * CI1, CI1, 0, stream>>>(EN2, ATT1, CI1, 1);

  // ---- 5) O = att * V  (bf16 x bf16 -> bf16), aliases Q buffers ----
  gemm_wmma<EPI_AV, true, true, false>
      <<<dim3(CI0 / 128, HW0 / 64, 16), blk, 0, stream>>>(
      ATT0, V0, V0, O0, nullptr, nullptr, nullptr, nullptr, 1.0f,
      CI0, HW0, CI0, CI0, HW0,
      0, (long long)CI0 * CI0, 0, (long long)CI0 * HW0,
      (long long)CI0 * HW0, 0, 0, 0);
  gemm_wmma<EPI_AV, true, true, false>
      <<<dim3(CI1 / 128, HW1 / 64, 16), blk, 0, stream>>>(
      ATT1, V1, V1, O1, nullptr, nullptr, nullptr, nullptr, 1.0f,
      CI1, HW1, CI1, CI1, HW1,
      0, (long long)CI1 * CI1, 0, (long long)CI1 * HW1,
      (long long)CI1 * HW1, 0, 0, 0);

  // ---- 6) Back-projection + residual: out = g*(rw*O + rb) + 2x ----
  // d_out order: ys00 | ys10 | ys01 | ys11
  float* out = (float*)d_out;
  const long long dD = (long long)NBATCH * (C0 * HW0 + C1 * HW1);  // 12582912
  gemm_wmma<EPI_OUT, false, true, false>
      <<<dim3(C0 / 128, HW0 / 64, 16), blk, 0, stream>>>(
      rw0, O0, O0, out, rb0, rgb0, freq0, g0, 1.0f,
      C0, HW0, CI0, CI0, HW0,
      (long long)C0 * CI0, 0, 0, (long long)CI0 * HW0,
      0, dD, (long long)C0 * HW0, (long long)C0 * HW0);
  gemm_wmma<EPI_OUT, false, true, false>
      <<<dim3(C1 / 128, HW1 / 64, 16), blk, 0, stream>>>(
      rw1, O1, O1, out + (long long)NBATCH * C0 * HW0, rb1, rgb1, freq1, g1, 1.0f,
      C1, HW1, CI1, CI1, HW1,
      (long long)C1 * CI1, 0, 0, (long long)CI1 * HW1,
      0, dD, (long long)C1 * HW1, (long long)C1 * HW1);
}
